// STPM_14628658610403
// MI455X (gfx1250) — matrix operand order repaired and checked by reading, not compiled, not run on hardware
//
#include <hip/hip_runtime.h>

// Problem constants (match the reference)
#define BB 8
#define PP 784
#define DD 1536
#define NN 20000
#define KK 9
#define ROWS (BB * PP)            // 6272 patch rows
#define ROW_TILES32 (ROWS / 32)   // 196 (32-row tiles)
#define COL_TILES (NN / 16)       // 1250
#define NSLICES 25
#define TILES_PER_SLICE (COL_TILES / NSLICES) // 50
#define PAIRS_PER_SLICE (TILES_PER_SLICE / 2) // 25
#define KSTEPS (DD / 32)          // 48

typedef __attribute__((ext_vector_type(16))) __bf16 bf16x16;
typedef __attribute__((ext_vector_type(8)))  float  f32x8;

union ABFrag { bf16x16 v; unsigned u[8]; uint4 q[2]; };

__device__ __forceinline__ unsigned short f2bf(float f) {
    unsigned u = __float_as_uint(f);
    unsigned r = u + 0x7FFFu + ((u >> 16) & 1u);   // round-to-nearest-even
    return (unsigned short)(r >> 16);
}

__device__ __forceinline__ unsigned packbf(float lo, float hi) {
    return (unsigned)f2bf(lo) | ((unsigned)f2bf(hi) << 16);
}

__device__ __forceinline__ void atomicMinF(float* addr, float val) {
    unsigned* ua = (unsigned*)addr;
    unsigned old = __float_as_uint(*addr);
    while (val < __uint_as_float(old)) {
        unsigned prev = atomicCAS(ua, old, __float_as_uint(val));
        if (prev == old) break;
        old = prev;
    }
}

// ---------------- K0: init closest_dist to +inf ----------------
__global__ void stpm_init(float* closest) {
    int i = blockIdx.x * blockDim.x + threadIdx.x;
    if (i < ROWS) closest[i] = 3.0e38f;
}

// ---- K1: fused row norms + fp32->bf16 conversion (one wave per row) ----
__global__ void stpm_prep(const float* __restrict__ mem,
                          const float* __restrict__ x,
                          float* __restrict__ m2, float* __restrict__ x2,
                          unsigned short* __restrict__ mb,
                          unsigned short* __restrict__ xb) {
    int gwave = (blockIdx.x * blockDim.x + threadIdx.x) >> 5;
    int lane  = threadIdx.x & 31;
    if (gwave >= NN + ROWS) return;                 // uniform per wave
    const float* src;
    float* dst;
    unsigned short* bdst;
    if (gwave < NN) {
        src = mem + (size_t)gwave * DD; dst = m2 + gwave;
        bdst = mb + (size_t)gwave * DD;
    } else {
        int r = gwave - NN;
        src = x + (size_t)r * DD; dst = x2 + r;
        bdst = xb + (size_t)r * DD;
    }
    const float2* s2 = (const float2*)src;
    unsigned* b2 = (unsigned*)bdst;
    float s = 0.f;
    for (int c = lane; c < DD / 2; c += 32) {
        float2 v = s2[c];
        s = fmaf(v.x, v.x, fmaf(v.y, v.y, s));
        b2[c] = packbf(v.x, v.y);
    }
    for (int off = 16; off; off >>= 1) s += __shfl_xor(s, off, 32);
    if (lane == 0) *dst = s;
}

// ---------------- K2: 32x32 BF16 WMMA GEMM + fused min-over-n ----------------
// 4 accumulators per wave; every A and B fragment feeds TWO wmmas ->
// 8 global_load_b128 + 4 v_wmma per K-step (2.0 mem-ops per wmma).
__global__ void __launch_bounds__(128)
stpm_gemm_min(const unsigned short* __restrict__ xb,
              const unsigned short* __restrict__ mb,
              const float* __restrict__ x2, const float* __restrict__ m2,
              float* __restrict__ closest) {
    const int rt   = blockIdx.x;              // 32-row tile 0..195
    const int sl   = blockIdx.y;              // column slice 0..24
    const int tid  = threadIdx.x;             // 0..127 (4 waves)
    const int wave = tid >> 5;
    const int lane = tid & 31;
    const int row0 = rt * 32;

    const int m  = lane & 15;                 // A: M index; B/C: N index
    const int hf = lane >> 4;                 // half-wave selector

    float rminL[8], rminH[8];
#pragma unroll
    for (int r = 0; r < 8; ++r) { rminL[r] = 3.0e38f; rminH[r] = 3.0e38f; }

    const int t0 = sl * TILES_PER_SLICE;
    for (int q = wave; q < PAIRS_PER_SLICE; q += 4) {
        const int n0 = (t0 + 2 * q) * 16;     // first column tile base
        const int nA = n0 + m;
        const int nB = n0 + 16 + m;

        // per-lane streaming pointers (fixed 64B stride per K-step)
        const unsigned short* pA0 = xb + (size_t)(row0 + m) * DD + hf * 8;
        const unsigned short* pA1 = pA0 + (size_t)16 * DD;   // rows row0+16..+31
        const unsigned short* pB0 = mb + (size_t)nA * DD + hf * 8;
        const unsigned short* pB1 = mb + (size_t)nB * DD + hf * 8;

        f32x8 acc00 = {}, acc01 = {}, acc10 = {}, acc11 = {};
#pragma unroll 2
        for (int ks = 0; ks < KSTEPS; ++ks) {
            ABFrag a0, a1, b0, b1;
            a0.q[0] = *(const uint4*)(pA0);       // x[m   ][k0+hf*8 .. +7]
            a0.q[1] = *(const uint4*)(pA0 + 16);  // x[m   ][k0+16+hf*8 .. +7]
            a1.q[0] = *(const uint4*)(pA1);       // x[m+16][..]
            a1.q[1] = *(const uint4*)(pA1 + 16);
            b0.q[0] = *(const uint4*)(pB0);       // mem[nA][k0+hf*8 .. +7]
            b0.q[1] = *(const uint4*)(pB0 + 16);  // mem[nA][k0+16+hf*8 .. +7]
            b1.q[0] = *(const uint4*)(pB1);
            b1.q[1] = *(const uint4*)(pB1 + 16);

            acc00 = __builtin_amdgcn_wmma_f32_16x16x32_bf16(
                false, a0.v, false, b0.v, (short)0, acc00, false, false);
            acc01 = __builtin_amdgcn_wmma_f32_16x16x32_bf16(
                false, a0.v, false, b1.v, (short)0, acc01, false, false);
            acc10 = __builtin_amdgcn_wmma_f32_16x16x32_bf16(
                false, a1.v, false, b0.v, (short)0, acc10, false, false);
            acc11 = __builtin_amdgcn_wmma_f32_16x16x32_bf16(
                false, a1.v, false, b1.v, (short)0, acc11, false, false);

            pA0 += 32; pA1 += 32; pB0 += 32; pB1 += 32;  // one K-step (64B)
        }

        // d(m',n) = x2[m'] + m2[n] - 2*acc ; fold x2 once at the very end.
        const float m2a = m2[nA];
        const float m2b = m2[nB];
#pragma unroll
        for (int r = 0; r < 8; ++r) {
            float ddL = fminf(fmaf(-2.0f, acc00[r], m2a),
                              fmaf(-2.0f, acc01[r], m2b));
            float ddH = fminf(fmaf(-2.0f, acc10[r], m2a),
                              fmaf(-2.0f, acc11[r], m2b));
            for (int off = 8; off; off >>= 1) {
                ddL = fminf(ddL, __shfl_xor(ddL, off, 16));
                ddH = fminf(ddH, __shfl_xor(ddH, off, 16));
            }
            rminL[r] = fminf(rminL[r], ddL);
            rminH[r] = fminf(rminH[r], ddH);
        }
    }

    // lanes 0 and 16 hold mins for M = r and M = 8 + r (within each 16-row block)
    if (m == 0) {
        const int mbse = hf * 8;
#pragma unroll
        for (int r = 0; r < 8; ++r) {
            int rowL = row0 + mbse + r;
            int rowH = row0 + 16 + mbse + r;
            atomicMinF(&closest[rowL], rminL[r] + x2[rowL]);
            atomicMinF(&closest[rowH], rminH[r] + x2[rowH]);
        }
    }
}

// ---------------- K3: per-batch argmax patch, exact top-9, score ----------------
__global__ void __launch_bounds__(256)
stpm_finalize(const float* __restrict__ x, const float* __restrict__ mem,
              const float* __restrict__ x2, const float* __restrict__ m2,
              const float* __restrict__ closest, float* __restrict__ distbuf,
              float* __restrict__ out) {
    const int b = blockIdx.x, tid = threadIdx.x;
    __shared__ float sval[256];
    __shared__ int   sidx[256];
    __shared__ int   pstar;
    __shared__ float maxclosest;
    __shared__ float sx[DD];
    __shared__ float topk[KK];

    // argmax over the 784 closest-distances of this batch (first-max tiebreak)
    float best = -3.0e38f; int bi = 0;
    for (int p = tid; p < PP; p += 256) {
        float v = closest[b * PP + p];
        if (v > best) { best = v; bi = p; }
    }
    sval[tid] = best; sidx[tid] = bi;
    __syncthreads();
    for (int s = 128; s; s >>= 1) {
        if (tid < s) {
            if (sval[tid + s] > sval[tid] ||
                (sval[tid + s] == sval[tid] && sidx[tid + s] < sidx[tid])) {
                sval[tid] = sval[tid + s]; sidx[tid] = sidx[tid + s];
            }
        }
        __syncthreads();
    }
    if (tid == 0) { pstar = sidx[0]; maxclosest = sval[0]; }
    __syncthreads();

    // stage the worst patch's embedding
    const size_t xoff = ((size_t)b * PP + pstar) * DD;
    for (int c = tid; c < DD; c += 256) sx[c] = x[xoff + c];
    __syncthreads();
    const float x2p = x2[b * PP + pstar];

    // fp32 distances to all 20000 memory rows: one wave per candidate
    const int wave = tid >> 5, lane = tid & 31;
    float* db = distbuf + (size_t)b * NN;
    for (int n = wave; n < NN; n += 8) {
        const float* mr = mem + (size_t)n * DD;
        float s = 0.f;
        for (int c = lane; c < DD; c += 32) s = fmaf(sx[c], mr[c], s);
        for (int off = 16; off; off >>= 1) s += __shfl_xor(s, off, 32);
        if (lane == 0) db[n] = x2p + m2[n] - 2.0f * s;
    }
    __syncthreads();

    // extract 9 smallest (ascending) by iterative argmin
    for (int it = 0; it < KK; ++it) {
        float bv = 3.0e38f; int bn = 0;
        for (int n = tid; n < NN; n += 256) {
            float v = db[n];
            if (v < bv) { bv = v; bn = n; }
        }
        sval[tid] = bv; sidx[tid] = bn;
        __syncthreads();
        for (int s = 128; s; s >>= 1) {
            if (tid < s) {
                if (sval[tid + s] < sval[tid] ||
                    (sval[tid + s] == sval[tid] && sidx[tid + s] < sidx[tid])) {
                    sval[tid] = sval[tid + s]; sidx[tid] = sidx[tid + s];
                }
            }
            __syncthreads();
        }
        if (tid == 0) { topk[it] = sval[0]; db[sidx[0]] = 3.0e38f; }
        __syncthreads();
    }

    if (tid == 0) {
        float mx = topk[0];
        for (int i = 1; i < KK; ++i) mx = fmaxf(mx, topk[i]);
        float sum = 0.f, mxe = 0.f;
        for (int i = 0; i < KK; ++i) {
            float e = __expf(topk[i] - mx);
            sum += e; mxe = fmaxf(mxe, e);
        }
        out[b] = (1.0f - mxe / sum) * maxclosest;
    }
}

extern "C" void kernel_launch(void* const* d_in, const int* in_sizes, int n_in,
                              void* d_out, int out_size, void* d_ws, size_t ws_size,
                              hipStream_t stream) {
    (void)in_sizes; (void)n_in; (void)out_size; (void)ws_size;
    const float* x   = (const float*)d_in[0];
    const float* mem = (const float*)d_in[1];
    float* out = (float*)d_out;

    // workspace layout:
    //   floats : closest[6272] | m2[20000] | x2[6272] | dist[8*20000]
    //   bf16   : mb[20000*1536] | xb[6272*1536]
    float* w       = (float*)d_ws;
    float* closest = w;
    float* m2      = closest + ROWS;
    float* x2      = m2 + NN;
    float* distbuf = x2 + ROWS;
    unsigned short* mb = (unsigned short*)(distbuf + (size_t)BB * NN);
    unsigned short* xb = mb + (size_t)NN * DD;

    stpm_init<<<(ROWS + 255) / 256, 256, 0, stream>>>(closest);

    const int nwaves = NN + ROWS;
    stpm_prep<<<(nwaves * 32 + 255) / 256, 256, 0, stream>>>(mem, x, m2, x2, mb, xb);

    stpm_gemm_min<<<dim3(ROW_TILES32, NSLICES), 128, 0, stream>>>(xb, mb, x2, m2, closest);

    stpm_finalize<<<BB, 256, 0, stream>>>(x, mem, x2, m2, closest, distbuf, out);
}
